// SVDNet_25563645346637
// MI455X (gfx1250) — matrix operand order repaired and checked by reading, not compiled, not run on hardware
//
#include <hip/hip_runtime.h>
#include <hip/hip_bf16.h>
#include <math.h>

typedef __attribute__((ext_vector_type(16))) _Float16 v16h;
typedef __attribute__((ext_vector_type(8)))  _Float16 v8h;
typedef __attribute__((ext_vector_type(8)))  float    v8f;

#define HW1 512
#define R_  128

// ---------------------------------------------------------------------------
// conv1: direct 2->32, 3x3, pad 1, relu, write f16 planes [oc][y][x]
// ---------------------------------------------------------------------------
__global__ __launch_bounds__(256) void conv1_kernel(
    const float* __restrict__ x, const float* __restrict__ w,
    const float* __restrict__ b, _Float16* __restrict__ out)
{
  __shared__ float ws[576];
  __shared__ float bs[32];
  int tid = threadIdx.x;
  for (int i = tid; i < 576; i += 256) ws[i] = w[i];
  if (tid < 32) bs[tid] = b[tid];
  __syncthreads();

  int idx = blockIdx.x * 256 + tid;
  int yy = idx >> 9;
  int xx = idx & 511;

  const float2* __restrict__ px = (const float2*)x;  // (y,x,2) channel pairs
  float pix[2][3][3];
  #pragma unroll
  for (int dy = 0; dy < 3; ++dy)
    #pragma unroll
    for (int dx = 0; dx < 3; ++dx) {
      int gy = yy + dy - 1, gx = xx + dx - 1;
      bool ok = (gy >= 0 && gy < HW1 && gx >= 0 && gx < HW1);
      float2 p = make_float2(0.f, 0.f);
      if (ok) p = px[((size_t)gy << 9) + gx];
      pix[0][dy][dx] = p.x; pix[1][dy][dx] = p.y;
    }

  for (int oc = 0; oc < 32; ++oc) {
    float acc = bs[oc];
    #pragma unroll
    for (int ic = 0; ic < 2; ++ic)
      #pragma unroll
      for (int dy = 0; dy < 3; ++dy)
        #pragma unroll
        for (int dx = 0; dx < 3; ++dx)
          acc += ws[oc * 18 + ic * 9 + dy * 3 + dx] * pix[ic][dy][dx];
    acc = acc > 0.f ? acc : 0.f;
    out[(size_t)oc * (HW1 * HW1) + idx] = (_Float16)acc;
  }
}

// ---------------------------------------------------------------------------
// f32 -> f16 convert (for conv2/conv3 weights)
// ---------------------------------------------------------------------------
__global__ void cvt_f16_kernel(const float* __restrict__ s, _Float16* __restrict__ d, int n)
{
  int i = blockIdx.x * 256 + threadIdx.x;
  if (i < n) d[i] = (_Float16)s[i];
}

// ---------------------------------------------------------------------------
// WMMA implicit-GEMM 3x3 conv, pad 1, + bias + relu.
//   A = weights (OC x K), K = IC*9, k = ic*9 + dy*3 + dx (== flat OIHW order).
//       Per-lane A fragment = two contiguous 8-half runs -> 2x global b128.
//   B = LDS tile staged as THREE dx-shifted copies; each B element is a
//       scalar LDS read at [dx][ic][dy][col] (16 per fragment).
// One A fragment per wave per K-chunk feeds two WMMA tiles (same M-block).
// MEAN=false: store f16 activation planes.
// MEAN=true : reduce relu output over the spatial tile -> per-WG partial sums.
// ---------------------------------------------------------------------------
template <int IC, int OC, int HW, int XT, bool MEAN>
__global__ __launch_bounds__(256) void conv_wmma_kernel(
    const _Float16* __restrict__ in, const _Float16* __restrict__ wf,
    const float* __restrict__ bias, _Float16* __restrict__ out,
    float* __restrict__ partial)
{
  constexpr int K   = IC * 9;
  constexpr int NCH = K / 32;        // K chunks of 32
  constexpr int NB  = XT / 16;       // N blocks
  constexpr int SM  = 3 * IC * 3 * XT;  // shifted-copy halves
  constexpr int TSZ = MEAN ? (OC * XT) : 1;
  static_assert(K % 32 == 0, "K must be chunkable by 32");
  static_assert(K % 8 == 0, "A vector loads need 16B alignment");
  static_assert(NB == 2 || NB == 4, "tile pairing assumes NB in {2,4}");

  __shared__ _Float16 bsh[SM];   // [s][ic][dy][c] , c in [0,XT)
  __shared__ float    tile[TSZ];

  const int y   = blockIdx.y;
  const int x0  = blockIdx.x * XT;
  const int tid = threadIdx.x;

  // stage three shifted copies: bsh[((s*IC+ic)*3+dy)*XT + c] = in[ic][y+dy-1][x0+c+s-1]
  for (int i = tid; i < SM; i += 256) {
    int c    = i & (XT - 1);
    int rest = i / XT;
    int dy   = rest % 3;  rest /= 3;
    int ic   = rest & (IC - 1);
    int s    = rest / IC;
    int gy = y + dy - 1;
    int gx = x0 + c + s - 1;
    _Float16 v = (_Float16)0.f;
    if (gy >= 0 && gy < HW && gx >= 0 && gx < HW)
      v = in[(size_t)ic * HW * HW + (size_t)gy * HW + gx];
    bsh[i] = v;
  }
  __syncthreads();

  const int wave = tid >> 5;
  const int lane = tid & 31;
  const int lo   = lane & 15;
  const int hi   = lane >> 4;

  // wave's two tiles share one M-block
  const int mb  = (2 * wave) / NB;
  const int nb0 = (2 * wave) & (NB - 1);   // second tile is nb0+1

  // per-lane B row decomposition depends only on (kc, lane)
  v8f c0 = {0.f,0.f,0.f,0.f,0.f,0.f,0.f,0.f};
  v8f c1 = {0.f,0.f,0.f,0.f,0.f,0.f,0.f,0.f};

  const int oc_a = mb * 16 + lo;            // A row (M = lo for both lane halves)
  const _Float16* __restrict__ wrow = wf + (size_t)oc_a * K;

  for (int kc = 0; kc < NCH; ++kc) {
    // A fragment: two contiguous 8-half runs (16B aligned)
    int kbase = kc * 32 + 8 * hi;
    v8h alo = *(const v8h*)(wrow + kbase);        // K = kbase .. +7
    v8h ahi = *(const v8h*)(wrow + kbase + 16);   // K = kbase+16 .. +23
    v16h a = __builtin_shufflevector(alo, ahi, 0,1,2,3,4,5,6,7,8,9,10,11,12,13,14,15);

    // B fragments: lane = N column (lo), elements = K in same order as A
    v16h b0, b1;
    #pragma unroll
    for (int e = 0; e < 16; ++e) {
      int k  = kc * 32 + 8 * hi + (e & 7) + ((e >> 3) << 4);
      int ic = k / 9;
      int r9 = k - ic * 9;
      int dy = r9 / 3;
      int dx = r9 - dy * 3;
      int base = (((dx * IC) + ic) * 3 + dy) * XT + nb0 * 16 + lo;
      b0[e] = bsh[base];
      b1[e] = bsh[base + 16];
    }

    c0 = __builtin_amdgcn_wmma_f32_16x16x32_f16(false, a, false, b0,
                                                (short)0, c0, false, false);
    c1 = __builtin_amdgcn_wmma_f32_16x16x32_f16(false, a, false, b1,
                                                (short)0, c1, false, false);
  }

  // D layout: VGPR r -> M = r + 8*hi, N = lo
  #pragma unroll
  for (int r = 0; r < 8; ++r) {
    int oc = mb * 16 + r + 8 * hi;
    float bz = bias[oc];
    float va = c0[r] + bz; va = va > 0.f ? va : 0.f;
    float vb = c1[r] + bz; vb = vb > 0.f ? vb : 0.f;
    int xpa = nb0 * 16 + lo;
    int xpb = xpa + 16;
    if constexpr (MEAN) {
      tile[oc * XT + xpa] = va;
      tile[oc * XT + xpb] = vb;
    } else {
      out[(size_t)oc * HW * HW + (size_t)y * HW + (x0 + xpa)] = (_Float16)va;
      out[(size_t)oc * HW * HW + (size_t)y * HW + (x0 + xpb)] = (_Float16)vb;
    }
  }

  if constexpr (MEAN) {
    __syncthreads();
    if (tid < OC) {
      float s = 0.f;
      for (int xx = 0; xx < XT; ++xx) s += tile[tid * XT + xx];
      partial[(size_t)(blockIdx.y * gridDim.x + blockIdx.x) * OC + tid] = s;
    }
  }
}

// ---------------------------------------------------------------------------
// 2x2 max pool, f16 planes (64,512,512) -> (64,256,256)
// ---------------------------------------------------------------------------
__global__ __launch_bounds__(256) void maxpool_kernel(
    const _Float16* __restrict__ in, _Float16* __restrict__ out)
{
  int idx = blockIdx.x * 256 + threadIdx.x;        // over 64*256*256
  int cch = idx >> 16;
  int rem = idx & 65535;
  int y   = rem >> 8;
  int x   = rem & 255;
  size_t base = (size_t)cch * (512 * 512) + (size_t)(y * 2) * 512 + (x * 2);
  float a0 = (float)in[base];
  float a1 = (float)in[base + 1];
  float a2 = (float)in[base + 512];
  float a3 = (float)in[base + 513];
  float m = fmaxf(fmaxf(a0, a1), fmaxf(a2, a3));
  out[idx] = (_Float16)m;
}

// ---------------------------------------------------------------------------
// feat[oc] = mean over spatial = (sum of partials, fixed order) / 65536
// ---------------------------------------------------------------------------
__global__ void feat_reduce_kernel(const float* __restrict__ partial,
                                   float* __restrict__ feat, int nwg)
{
  int oc = threadIdx.x;   // 128 threads
  float s = 0.f;
  for (int w = 0; w < nwg; ++w) s += partial[(size_t)w * 128 + oc];
  feat[oc] = s * (1.0f / 65536.0f);
}

// ---------------------------------------------------------------------------
// Wave-per-row matvec: y = act(W @ x + b). act: 0 none, 1 relu, 2 softplus
// cols must be a multiple of 128 (true for all calls) -> float4, no tails.
// ---------------------------------------------------------------------------
__global__ __launch_bounds__(256) void matvec_kernel(
    const float* __restrict__ w, const float* __restrict__ bias,
    const float* __restrict__ x, float* __restrict__ y,
    int rows, int cols, int act)
{
  int wave = threadIdx.x >> 5;
  int lane = threadIdx.x & 31;
  int row  = blockIdx.x * 8 + wave;
  if (row >= rows) return;
  const float4* __restrict__ wr = (const float4*)(w + (size_t)row * cols);
  const float4* __restrict__ x4 = (const float4*)x;
  int n4 = cols >> 2;
  float s = 0.f;
  for (int i = lane; i < n4; i += 32) {
    float4 a = wr[i];
    float4 b = x4[i];
    s += a.x * b.x + a.y * b.y + a.z * b.z + a.w * b.w;
  }
  #pragma unroll
  for (int off = 16; off > 0; off >>= 1) s += __shfl_down(s, off, 32);
  if (lane == 0) {
    s += bias[row];
    if (act == 1)      s = s > 0.f ? s : 0.f;
    else if (act == 2) s = fmaxf(s, 0.f) + log1pf(expf(-fabsf(s)));
    y[row] = s;
  }
}

// ---------------------------------------------------------------------------
// 128-element bitonic sort (ascending in LDS), emit descending
// ---------------------------------------------------------------------------
__global__ __launch_bounds__(128) void sort_desc_kernel(
    const float* __restrict__ s, float* __restrict__ out)
{
  __shared__ float sd[128];
  int tid = threadIdx.x;
  sd[tid] = s[tid];
  __syncthreads();
  for (int k = 2; k <= 128; k <<= 1) {
    for (int j = k >> 1; j > 0; j >>= 1) {
      int ixj = tid ^ j;
      if (ixj > tid) {
        float a = sd[tid], b = sd[ixj];
        bool asc = ((tid & k) == 0);
        bool swap = asc ? (a > b) : (a < b);
        if (swap) { sd[tid] = b; sd[ixj] = a; }
      }
      __syncthreads();
    }
  }
  out[tid] = sd[127 - tid];
}

// ---------------------------------------------------------------------------
// Modified Gram-Schmidt on (512,128) complex, stored (m, r, 2).
// One 512-thread WG per matrix; thread = row; Q lives in dst (global).
// Matches reference: v_j -= <q_i, v_j> q_i sequentially, norm = sqrt(|v|^2+1e-8)
// ---------------------------------------------------------------------------
__global__ __launch_bounds__(512) void gram_kernel(
    const float* __restrict__ Ur, const float* __restrict__ Vr,
    float* __restrict__ outU, float* __restrict__ outV)
{
  const float* src = (blockIdx.x == 0) ? Ur : Vr;
  float*       dst = (blockIdx.x == 0) ? outU : outV;

  __shared__ float redA[16], redB[16];
  __shared__ float bc[2];

  const int tid  = threadIdx.x;   // row m
  const int lane = tid & 31;
  const int wave = tid >> 5;

  for (int j = 0; j < R_; ++j) {
    float vre = src[(size_t)tid * (R_ * 2) + j * 2];
    float vim = src[(size_t)tid * (R_ * 2) + j * 2 + 1];

    for (int i = 0; i < j; ++i) {
      float qre = dst[(size_t)tid * (R_ * 2) + i * 2];
      float qim = dst[(size_t)tid * (R_ * 2) + i * 2 + 1];
      // c = <q_i, v> = sum conj(q)*v
      float pa = qre * vre + qim * vim;
      float pb = qre * vim - qim * vre;
      #pragma unroll
      for (int off = 16; off > 0; off >>= 1) {
        pa += __shfl_down(pa, off, 32);
        pb += __shfl_down(pb, off, 32);
      }
      if (lane == 0) { redA[wave] = pa; redB[wave] = pb; }
      __syncthreads();
      if (tid == 0) {
        float sa = 0.f, sb = 0.f;
        for (int q = 0; q < 16; ++q) { sa += redA[q]; sb += redB[q]; }
        bc[0] = sa; bc[1] = sb;
      }
      __syncthreads();
      float cre = bc[0], cim = bc[1];
      vre -= cre * qre - cim * qim;
      vim -= cre * qim + cim * qre;
    }

    // norm
    float pa = vre * vre + vim * vim;
    #pragma unroll
    for (int off = 16; off > 0; off >>= 1) pa += __shfl_down(pa, off, 32);
    if (lane == 0) redA[wave] = pa;
    __syncthreads();
    if (tid == 0) {
      float sa = 0.f;
      for (int q = 0; q < 16; ++q) sa += redA[q];
      bc[0] = sa;
    }
    __syncthreads();
    float invn = 1.0f / sqrtf(bc[0] + 1e-8f);
    dst[(size_t)tid * (R_ * 2) + j * 2]     = vre * invn;
    dst[(size_t)tid * (R_ * 2) + j * 2 + 1] = vim * invn;
    __syncthreads();  // protect redA/bc for next column
  }
}

// ---------------------------------------------------------------------------
// host-side launch
// ---------------------------------------------------------------------------
extern "C" void kernel_launch(void* const* d_in, const int* in_sizes, int n_in,
                              void* d_out, int out_size, void* d_ws, size_t ws_size,
                              hipStream_t stream) {
  (void)in_sizes; (void)n_in; (void)out_size; (void)ws_size;

  const float* x       = (const float*)d_in[0];
  const float* conv_w1 = (const float*)d_in[1];
  const float* conv_b1 = (const float*)d_in[2];
  const float* conv_w2 = (const float*)d_in[3];
  const float* conv_b2 = (const float*)d_in[4];
  const float* conv_w3 = (const float*)d_in[5];
  const float* conv_b3 = (const float*)d_in[6];
  const float* fc_w1   = (const float*)d_in[7];
  const float* fc_b1   = (const float*)d_in[8];
  const float* fc_w2   = (const float*)d_in[9];
  const float* fc_b2   = (const float*)d_in[10];
  const float* s_w1    = (const float*)d_in[11];
  const float* s_b1    = (const float*)d_in[12];
  const float* s_w2    = (const float*)d_in[13];
  const float* s_b2    = (const float*)d_in[14];
  const float* u_w1    = (const float*)d_in[15];
  const float* u_b1    = (const float*)d_in[16];
  const float* u_w2    = (const float*)d_in[17];
  const float* u_b2    = (const float*)d_in[18];
  const float* v_w1    = (const float*)d_in[19];
  const float* v_b1    = (const float*)d_in[20];
  const float* v_w2    = (const float*)d_in[21];
  const float* v_b2    = (const float*)d_in[22];

  // workspace carve-up (bytes, 256B aligned)
  char* ws = (char*)d_ws;
  size_t off = 0;
  auto alloc = [&](size_t bytes) {
    char* p = ws + off;
    off += (bytes + 255) & ~(size_t)255;
    return p;
  };
  _Float16* c1out  = (_Float16*)alloc((size_t)32  * 512 * 512 * 2);  // 16.8 MB
  _Float16* c2out  = (_Float16*)alloc((size_t)64  * 512 * 512 * 2);  // 33.6 MB
  _Float16* pool   = (_Float16*)alloc((size_t)64  * 256 * 256 * 2);  //  8.4 MB
  _Float16* w2h    = (_Float16*)alloc((size_t)64  * 288 * 2);
  _Float16* w3h    = (_Float16*)alloc((size_t)128 * 576 * 2);
  float* partial   = (float*)alloc((size_t)2048 * 128 * 4);          //  1 MB
  float* feat      = (float*)alloc(128 * 4);
  float* h1        = (float*)alloc(256 * 4);
  float* fvec      = (float*)alloc(512 * 4);
  float* t1        = (float*)alloc(256 * 4);
  float* sraw      = (float*)alloc(128 * 4);
  float* uh        = (float*)alloc(512 * 4);
  float* vh        = (float*)alloc(512 * 4);
  float* Uraw      = (float*)alloc((size_t)512 * 128 * 2 * 4);       // 0.5 MB
  float* Vraw      = (float*)alloc((size_t)512 * 128 * 2 * 4);       // 0.5 MB

  float* outQu = (float*)d_out;                    // 512*128*2
  float* outS  = (float*)d_out + 512 * 128 * 2;    // 128
  float* outQv = outS + 128;                       // 512*128*2

  // conv stack
  conv1_kernel<<<1024, 256, 0, stream>>>(x, conv_w1, conv_b1, c1out);
  cvt_f16_kernel<<<(18432 + 255) / 256, 256, 0, stream>>>(conv_w2, w2h, 18432);
  cvt_f16_kernel<<<(73728 + 255) / 256, 256, 0, stream>>>(conv_w3, w3h, 73728);

  conv_wmma_kernel<32, 64, 512, 64, false>
      <<<dim3(8, 512), 256, 0, stream>>>(c1out, w2h, conv_b2, c2out, nullptr);
  maxpool_kernel<<<16384, 256, 0, stream>>>(c2out, pool);
  conv_wmma_kernel<64, 128, 256, 32, true>
      <<<dim3(8, 256), 256, 0, stream>>>(pool, w3h, conv_b3, nullptr, partial);
  feat_reduce_kernel<<<1, 128, 0, stream>>>(partial, feat, 2048);

  // fc trunk
  matvec_kernel<<<(256 + 7) / 8, 256, 0, stream>>>(fc_w1, fc_b1, feat, h1, 256, 128, 1);
  matvec_kernel<<<(512 + 7) / 8, 256, 0, stream>>>(fc_w2, fc_b2, h1, fvec, 512, 256, 1);

  // s head
  matvec_kernel<<<(256 + 7) / 8, 256, 0, stream>>>(s_w1, s_b1, fvec, t1, 256, 512, 1);
  matvec_kernel<<<(128 + 7) / 8, 256, 0, stream>>>(s_w2, s_b2, t1, sraw, 128, 256, 2);

  // u head
  matvec_kernel<<<(512 + 7) / 8, 256, 0, stream>>>(u_w1, u_b1, fvec, uh, 512, 512, 1);
  matvec_kernel<<<(131072 + 7) / 8, 256, 0, stream>>>(u_w2, u_b2, uh, Uraw, 131072, 512, 0);

  // v head
  matvec_kernel<<<(512 + 7) / 8, 256, 0, stream>>>(v_w1, v_b1, fvec, vh, 512, 512, 1);
  matvec_kernel<<<(131072 + 7) / 8, 256, 0, stream>>>(v_w2, v_b2, vh, Vraw, 131072, 512, 0);

  // sort + orthogonalize
  sort_desc_kernel<<<1, 128, 0, stream>>>(sraw, outS);
  gram_kernel<<<2, 512, 0, stream>>>(Uraw, Vraw, outQu, outQv);
}
